// MultiheadAttention_36404142801137
// MI455X (gfx1250) — matrix operand order repaired and checked by reading, not compiled, not run on hardware
//
#include <hip/hip_runtime.h>
#include <hip/hip_bf16.h>

// MI455X / gfx1250: wave32, WMMA 16x16x32 f16 -> f32 accumulate.
// All LDS tiles are stored in WMMA *fragment order* so each fragment load is
// a single 32B LDS access (2x ds_load_b128) instead of 16 scalar gathers.

typedef __attribute__((ext_vector_type(16))) _Float16 v16h;
typedef __attribute__((ext_vector_type(8)))  _Float16 v8h;
typedef __attribute__((ext_vector_type(2)))  __fp16   pk2;  // cvt_pkrtz result
typedef __attribute__((ext_vector_type(8)))  float    v8f;
typedef __attribute__((ext_vector_type(4)))  float    v4f;

#define T_ 2048
#define B_ 2
#define E_ 1024
#define H_ 16
#define D_ 64

// ---------------- fragment-order offsets (ISA 7.12.2 layouts) ---------------
// A 16x32 f16 fragment block = 512 halves; lane (m&15)+16*hi owns 16
// consecutive halves. Element (m,k): hi=(k>>3)&1, j=((k&7)>>1)+4*(k>>4).
__device__ __forceinline__ int a_off(int m, int k) {
  int hi = (k >> 3) & 1;
  int j  = ((k & 7) >> 1) + ((k >> 4) << 2);
  return (((m & 15) + (hi << 4)) << 4) + (j << 1) + (k & 1);
}
// B 32x16 f16 fragment block = 512 halves; element (k,n): hi=(k>>3)&1, j=((k&7)>>1)+4*(k>>4).
__device__ __forceinline__ int b_off(int k, int n) {
  int hi = (k >> 3) & 1;
  int j  = ((k & 7) >> 1) + ((k >> 4) << 2);
  return (((n & 15) + (hi << 4)) << 4) + (j << 1) + (k & 1);
}

// One fragment = this lane's 32B slice of a 512-half block (2x ds_load_b128).
__device__ __forceinline__ v16h frag_ld(const _Float16* p, int lane) {
  return *(const v16h*)(p + (lane << 4));
}

// 8 f32 -> 8 packed f16 (v_cvt_pk_rtz_f16_f32) -> one 16B LDS store.
__device__ __forceinline__ void cvt_store8(_Float16* dst, v4f a, v4f b) {
  union { v8h v; pk2 h[4]; } u;
  u.h[0] = __builtin_amdgcn_cvt_pkrtz(a.x, a.y);
  u.h[1] = __builtin_amdgcn_cvt_pkrtz(a.z, a.w);
  u.h[2] = __builtin_amdgcn_cvt_pkrtz(b.x, b.y);
  u.h[3] = __builtin_amdgcn_cvt_pkrtz(b.z, b.w);
  *(v8h*)dst = u.v;
}

__device__ __forceinline__ v8f wmma32(v16h a, v16h b, v8f c) {
  return __builtin_amdgcn_wmma_f32_16x16x32_f16(false, a, false, b, (short)0, c,
                                                false, false);
}

// ---------------- Kernel 1: QKV projection, fp32 in -> f16 head layout ------
// Out[b][h][t][d] = X @ W^T + bias ; X:(M=T*B, E) f32, W:(E,E) f32 row-major.
__global__ __launch_bounds__(256) void proj_gemm(
    const float* __restrict__ X, const float* __restrict__ W,
    const float* __restrict__ bias, _Float16* __restrict__ Out) {
  __shared__ __align__(16) _Float16 As[8 * 512];  // 8 row-subtiles, frag order
  __shared__ __align__(16) _Float16 Bs[4 * 512];  // 4 col-subtiles, frag order
  const int tid = threadIdx.x, lane = tid & 31, wv = tid >> 5;
  const int r = lane & 15, hi = lane >> 4;
  const int n0 = blockIdx.x * 64, m0 = blockIdx.y * 128;
  v8f acc[4] = {};

  for (int k0 = 0; k0 < E_; k0 += 32) {
    // A: 128 rows x 4 k-chunks(8) = 512 chunks; 32B global -> 16B LDS each.
    for (int c = tid; c < 512; c += 256) {
      int m = c >> 2, kc = (c & 3) << 3;
      const float* src = &X[(size_t)(m0 + m) * E_ + k0 + kc];
      cvt_store8(&As[((m >> 4) << 9) + a_off(m & 15, kc)],
                 *(const v4f*)src, *(const v4f*)(src + 4));
    }
    // B: Bfrag[k][n] = W[n0+n][k0+k]; 64 n x 4 k-chunks = 256 chunks.
    {
      int c = tid;
      int n = c >> 2, kc = (c & 3) << 3;
      const float* src = &W[(size_t)(n0 + n) * E_ + k0 + kc];
      cvt_store8(&Bs[((n >> 4) << 9) + b_off(kc, n & 15)],
                 *(const v4f*)src, *(const v4f*)(src + 4));
    }
    __syncthreads();
    v16h a = frag_ld(&As[wv << 9], lane);
#pragma unroll
    for (int ns = 0; ns < 4; ++ns)
      acc[ns] = wmma32(a, frag_ld(&Bs[ns << 9], lane), acc[ns]);
    __syncthreads();
  }
#pragma unroll
  for (int ns = 0; ns < 4; ++ns) {
#pragma unroll
    for (int i = 0; i < 8; ++i) {
      int m = m0 + wv * 16 + i + 8 * hi;  // m = t*B + b
      int n = n0 + ns * 16 + r;           // n = h*64 + d
      float v = acc[ns][i] + bias[n];
      int t = m >> 1, b = m & 1;          // B_ == 2
      int h = n >> 6, d = n & 63;
      Out[(((size_t)b * H_ + h) * T_ + t) * D_ + d] = (_Float16)v;
    }
  }
}

// ---------------- Kernel 2: flash attention per (b,h), 64-row q tile --------
__global__ __launch_bounds__(128) void attn_kernel(
    const _Float16* __restrict__ Qh, const _Float16* __restrict__ Kh,
    const _Float16* __restrict__ Vh, const unsigned char* __restrict__ mask,
    _Float16* __restrict__ Ctx) {
  __shared__ __align__(16) _Float16 Qs[4 * 2 * 512];  // [wv][kfrag] A-frag order
  __shared__ __align__(16) _Float16 Ks[2 * 2 * 512];  // [ns][kfrag] B-frag order
  __shared__ __align__(16) _Float16 Vs[4 * 512];      // [dsub]      B-frag order
  __shared__ __align__(16) _Float16 Ps[4 * 512];      // per-wave P, A-frag order
  const int tid = threadIdx.x, lane = tid & 31, wv = tid >> 5;
  const int r = lane & 15, hi = lane >> 4;
  const int q0 = blockIdx.x * 64;
  const int bh = blockIdx.y;               // bh = b*H + h
  const int b = bh / H_, h = bh % H_;
  const size_t hb = (size_t)bh * T_ * D_;

  // Q tile 64x64 -> A-frag order: 64 rows x 8 k-chunks = 512 (f16 passthrough)
  for (int c = tid; c < 512; c += 128) {
    int m = c >> 3, kc = (c & 7) << 3;
    v8h q = *(const v8h*)&Qh[hb + (size_t)(q0 + m) * D_ + kc];
    *(v8h*)&Qs[((m >> 4) << 10) + ((kc >> 5) << 9) + a_off(m & 15, kc & 31)] = q;
  }
  __syncthreads();
  const v16h qa0 = frag_ld(&Qs[wv << 10], lane);
  const v16h qa1 = frag_ld(&Qs[(wv << 10) + 512], lane);

  float run_m[8], run_l[8];
  v8f o[4] = {};
#pragma unroll
  for (int i = 0; i < 8; ++i) { run_m[i] = -3.0e38f; run_l[i] = 0.f; }

  for (int s0 = 0; s0 < T_; s0 += 32) {
    // K tile: Bfrag[k=d][n=s-col]; 32 s x 8 d-chunks = 256 chunks.
    for (int c = tid; c < 256; c += 128) {
      int n = c >> 3, kc = (c & 7) << 3;
      v8h kvv = *(const v8h*)&Kh[hb + (size_t)(s0 + n) * D_ + kc];
      *(v8h*)&Ks[((n >> 4) << 10) + ((kc >> 5) << 9) + b_off(kc & 31, n & 15)] = kvv;
    }
    // V tile: Bfrag[k=s-row][n=d]; N contiguous in memory -> b16 scatter.
    for (int c = tid; c < 256; c += 128) {
      int s = c >> 3, dc = (c & 7) << 3;
      v8h vv = *(const v8h*)&Vh[hb + (size_t)(s0 + s) * D_ + dc];
#pragma unroll
      for (int t = 0; t < 8; ++t) {
        int d = dc + t;
        Vs[((d >> 4) << 9) + b_off(s, d & 15)] = vv[t];
      }
    }
    if (s0 + 32 < T_ && tid < 32) {  // global_prefetch_b8 next tile
      __builtin_prefetch(&Kh[hb + (size_t)(s0 + 32 + tid) * D_], 0, 1);
      __builtin_prefetch(&Vh[hb + (size_t)(s0 + 32 + tid) * D_], 0, 1);
    }
    __syncthreads();

    // scores: Q(16x64) @ K^T(64x32) for this wave's 16 rows
    v8f cs[2];
#pragma unroll
    for (int ns = 0; ns < 2; ++ns) {
      v8f c = {};
      c = wmma32(qa0, frag_ld(&Ks[ns << 10], lane), c);
      c = wmma32(qa1, frag_ld(&Ks[(ns << 10) + 512], lane), c);
      int scol = s0 + ns * 16 + r;
      bool mk = mask[(size_t)b * T_ + scol] != 0;
#pragma unroll
      for (int i = 0; i < 8; ++i) c[i] = mk ? -3.0e38f : c[i] * 0.125f;
      cs[ns] = c;
    }

    // online softmax; each C-frag row is striped over a 16-lane half
#pragma unroll
    for (int i = 0; i < 8; ++i) {
      float mt = fmaxf(cs[0][i], cs[1][i]);
      mt = fmaxf(mt, __shfl_xor(mt, 1, 32));
      mt = fmaxf(mt, __shfl_xor(mt, 2, 32));
      mt = fmaxf(mt, __shfl_xor(mt, 4, 32));
      mt = fmaxf(mt, __shfl_xor(mt, 8, 32));
      float mnew = fmaxf(run_m[i], mt);
      float alpha = __expf(run_m[i] - mnew);
      float p0 = __expf(cs[0][i] - mnew);
      float p1 = __expf(cs[1][i] - mnew);
      cs[0][i] = p0; cs[1][i] = p1;
      float sum = p0 + p1;
      sum += __shfl_xor(sum, 1, 32);
      sum += __shfl_xor(sum, 2, 32);
      sum += __shfl_xor(sum, 4, 32);
      sum += __shfl_xor(sum, 8, 32);
      run_l[i] = run_l[i] * alpha + sum;
      run_m[i] = mnew;
#pragma unroll
      for (int d = 0; d < 4; ++d) o[d][i] *= alpha;
    }

    // P: C-layout -> A-frag order in this wave's private LDS block
#pragma unroll
    for (int i = 0; i < 8; ++i) {
      Ps[(wv << 9) + a_off(i + 8 * hi, r)]      = (_Float16)cs[0][i];
      Ps[(wv << 9) + a_off(i + 8 * hi, 16 + r)] = (_Float16)cs[1][i];
    }
    asm volatile("s_wait_dscnt 0" ::: "memory");  // in-wave LDS RAW
    v16h pa = frag_ld(&Ps[wv << 9], lane);
#pragma unroll
    for (int d = 0; d < 4; ++d)
      o[d] = wmma32(pa, frag_ld(&Vs[d << 9], lane), o[d]);
    __syncthreads();
  }

  // ctx[(t*B+b)*E + h*64 + d]
#pragma unroll
  for (int d = 0; d < 4; ++d) {
#pragma unroll
    for (int i = 0; i < 8; ++i) {
      int t = q0 + wv * 16 + i + 8 * hi;
      int e = h * 64 + d * 16 + r;
      float v = o[d][i] / fmaxf(run_l[i], 1e-30f);
      Ctx[((size_t)t * B_ + b) * E_ + e] = (_Float16)v;
    }
  }
}

// ---------------- Kernel 3: output projection, f16 ctx -> f32 out -----------
__global__ __launch_bounds__(256) void outproj_gemm(
    const _Float16* __restrict__ X, const float* __restrict__ W,
    const float* __restrict__ bias, float* __restrict__ Out) {
  __shared__ __align__(16) _Float16 As[8 * 512];
  __shared__ __align__(16) _Float16 Bs[4 * 512];
  const int tid = threadIdx.x, lane = tid & 31, wv = tid >> 5;
  const int r = lane & 15, hi = lane >> 4;
  const int n0 = blockIdx.x * 64, m0 = blockIdx.y * 128;
  v8f acc[4] = {};

  for (int k0 = 0; k0 < E_; k0 += 32) {
    for (int c = tid; c < 512; c += 256) {  // A: f16 passthrough, frag order
      int m = c >> 2, kc = (c & 3) << 3;
      v8h x = *(const v8h*)&X[(size_t)(m0 + m) * E_ + k0 + kc];
      *(v8h*)&As[((m >> 4) << 9) + a_off(m & 15, kc)] = x;
    }
    {
      int c = tid;
      int n = c >> 2, kc = (c & 3) << 3;
      const float* src = &W[(size_t)(n0 + n) * E_ + k0 + kc];
      cvt_store8(&Bs[((n >> 4) << 9) + b_off(kc, n & 15)],
                 *(const v4f*)src, *(const v4f*)(src + 4));
    }
    __syncthreads();
    v16h a = frag_ld(&As[wv << 9], lane);
#pragma unroll
    for (int ns = 0; ns < 4; ++ns)
      acc[ns] = wmma32(a, frag_ld(&Bs[ns << 9], lane), acc[ns]);
    __syncthreads();
  }
#pragma unroll
  for (int ns = 0; ns < 4; ++ns) {
#pragma unroll
    for (int i = 0; i < 8; ++i) {
      int m = m0 + wv * 16 + i + 8 * hi;
      int n = n0 + ns * 16 + r;
      Out[(size_t)m * E_ + n] = acc[ns][i] + bias[n];
    }
  }
}

// ---------------- launch -----------------------------------------------------
extern "C" void kernel_launch(void* const* d_in, const int* in_sizes, int n_in,
                              void* d_out, int out_size, void* d_ws,
                              size_t ws_size, hipStream_t stream) {
  const float* query = (const float*)d_in[0];
  const float* key   = (const float*)d_in[1];
  const float* value = (const float*)d_in[2];
  const unsigned char* kpm = (const unsigned char*)d_in[3];  // bool (B,T)
  const float* Wq = (const float*)d_in[4];
  const float* bq = (const float*)d_in[5];
  const float* Wk = (const float*)d_in[6];
  const float* bk = (const float*)d_in[7];
  const float* Wo = (const float*)d_in[8];
  const float* bo = (const float*)d_in[9];
  float* out = (float*)d_out;

  // workspace: qh,kh,vh (B*H*T*D f16 each) + ctx (T*B*E f16) = 32 MB total
  const size_t per = (size_t)B_ * H_ * T_ * D_;
  _Float16* qh  = (_Float16*)d_ws;
  _Float16* kh  = qh + per;
  _Float16* vh  = kh + per;
  _Float16* ctx = vh + per;
  (void)in_sizes; (void)n_in; (void)out_size; (void)ws_size;

  dim3 pgrid(E_ / 64, (T_ * B_) / 128);  // (16, 32)
  proj_gemm<<<pgrid, 256, 0, stream>>>(query, Wq, bq, qh);
  proj_gemm<<<pgrid, 256, 0, stream>>>(key, Wk, bk, kh);
  proj_gemm<<<pgrid, 256, 0, stream>>>(value, Wk, bk, vh);  // ref bug: v uses Wk/bk

  dim3 agrid(T_ / 64, B_ * H_);          // (32, 32)
  attn_kernel<<<agrid, 128, 0, stream>>>(qh, kh, vh, kpm, ctx);

  outproj_gemm<<<pgrid, 256, 0, stream>>>(ctx, Wo, bo, out);
}
